// SlidingWindowAttention_80960133530027
// MI455X (gfx1250) — compile-verified, matrix operand order checked by reading, not run on hardware
//
#include <hip/hip_runtime.h>

// ---------------------------------------------------------------------------
// Types for CDNA5 WMMA (wave32, 16x16x32 bf16 -> f32)
// ---------------------------------------------------------------------------
typedef __bf16 bf16;
typedef __attribute__((ext_vector_type(16))) __bf16 v16bf;
typedef __attribute__((ext_vector_type(8)))  float  v8f;
typedef __attribute__((ext_vector_type(4)))  unsigned int u32x4; // 16 bytes
typedef __attribute__((ext_vector_type(4)))  __bf16 v4bf;

__device__ __forceinline__ v8f wmma_bf16(v16bf a, v16bf b, v8f c) {
  // D = A(16x32) * B(32x16) + C, f32 accumulate
  return __builtin_amdgcn_wmma_f32_16x16x32_bf16(
      /*neg_a=*/false, a, /*neg_b=*/false, b,
      /*c_mod=*/(short)0, c, /*reuse_a=*/false, /*reuse_b=*/false);
}

// 16-element bf16 fragment for the 16-bit 16x32 A / 32x16 B operand.
// Lanes 0-15 hold K{0..7,16..23}; lanes 16-31 hold K{8..15,24..31}.
__device__ __forceinline__ v16bf load_frag_at(const bf16* rowptr, int kbase, int half) {
  union { u32x4 q[2]; v16bf v; } f;
  f.q[0] = *reinterpret_cast<const u32x4*>(rowptr + kbase + half * 8);
  f.q[1] = *reinterpret_cast<const u32x4*>(rowptr + kbase + 16 + half * 8);
  return f.v;
}

// gfx1250 async global->LDS copy, 16B per lane, tracked by ASYNCcnt.
__device__ __forceinline__ void async_b128(unsigned lds_off, const bf16* gptr) {
  asm volatile("global_load_async_to_lds_b128 %0, %1, off"
               :: "v"(lds_off), "v"(gptr) : "memory");
}
__device__ __forceinline__ void wait_async0() {
  asm volatile("s_wait_asynccnt 0x0" ::: "memory");
}
__device__ __forceinline__ unsigned lds_off_of(const void* p) {
  // LDS aperture keeps the LDS byte offset in addr[31:0]
  return (unsigned)(uintptr_t)p;
}

// ---------------------------------------------------------------------------
// Stage 0: f32 -> bf16 conversion
// ---------------------------------------------------------------------------
__global__ __launch_bounds__(256) void cvt_f32_bf16(const float* __restrict__ in,
                                                    bf16* __restrict__ out, int n) {
  int i = (blockIdx.x * blockDim.x + threadIdx.x) * 4;
  if (i >= n) return;
  float4 v = *reinterpret_cast<const float4*>(in + i);
  v4bf o;
  o[0] = (bf16)v.x; o[1] = (bf16)v.y; o[2] = (bf16)v.z; o[3] = (bf16)v.w;
  *reinterpret_cast<v4bf*>(out + i) = o;
}

// ---------------------------------------------------------------------------
// Stage 1/3: tiled bf16 GEMM  C[M,N] = A[M,K] * Bw[N,K]^T + bias[N]
//   MODE 0: store bf16 scattered into (B,H,L,D)   (QKV projections)
//   MODE 1: store f32 row-major M x N             (output projection)
// Block: 256 threads (8 waves). Block tile 128x256, BK=32, double-buffered
// LDS fed by async global->LDS. Wave tile 64x64 = 16 WMMAs per K-step.
// ---------------------------------------------------------------------------
template <int MODE>
__global__ __launch_bounds__(256) void gemm_bf16_kernel(
    const bf16* __restrict__ A, const bf16* __restrict__ Bw,
    const float* __restrict__ bias,
    bf16* __restrict__ out_qkv, float* __restrict__ out_f32,
    int M, int N, int K) {
  __shared__ bf16 As[2][128 * 32];
  __shared__ bf16 Bs[2][256 * 32];

  const int tid  = threadIdx.x;
  const int lane = tid & 31;
  const int wid  = tid >> 5;
  const int half = lane >> 4;
  const int l16  = lane & 15;
  const int bm = blockIdx.y * 128;
  const int bn = blockIdx.x * 256;
  const int wm = (wid >> 2) * 64;   // 2 row groups of 64
  const int wn = (wid & 3) * 64;    // 4 col groups of 64

  v8f acc[4][4];
  for (int i = 0; i < 4; ++i)
    for (int j = 0; j < 4; ++j)
      for (int e = 0; e < 8; ++e) acc[i][j][e] = 0.0f;

  // per-thread async-load assignment for one K-step (BK=32)
  const int arow = tid >> 1;             // 0..127
  const int acol = (tid & 1) * 16;       // 0 or 16
  const bf16* gA = A  + (size_t)(bm + arow) * K + acol;  // + k0
  const bf16* gB = Bw + (size_t)(bn + tid) * K;          // + k0 (one row/thread)
  const unsigned asOff[2] = { lds_off_of(&As[0][arow * 32 + acol]),
                              lds_off_of(&As[1][arow * 32 + acol]) };
  const unsigned bsOff[2] = { lds_off_of(&Bs[0][tid * 32]),
                              lds_off_of(&Bs[1][tid * 32]) };

  // prologue: fill buffer 0
  async_b128(asOff[0],      gA);
  async_b128(asOff[0] + 16, gA + 8);
  async_b128(bsOff[0],      gB);
  async_b128(bsOff[0] + 16, gB + 8);
  async_b128(bsOff[0] + 32, gB + 16);
  async_b128(bsOff[0] + 48, gB + 24);
  wait_async0();
  __syncthreads();

  for (int k0 = 0; k0 < K; k0 += 32) {
    const int cur = (k0 >> 5) & 1;
    const int nxt = cur ^ 1;
    if (k0 + 32 < K) {   // issue async loads for the next stage
      const bf16* nA = gA + k0 + 32;
      const bf16* nB = gB + k0 + 32;
      async_b128(asOff[nxt],      nA);
      async_b128(asOff[nxt] + 16, nA + 8);
      async_b128(bsOff[nxt],      nB);
      async_b128(bsOff[nxt] + 16, nB + 8);
      async_b128(bsOff[nxt] + 32, nB + 16);
      async_b128(bsOff[nxt] + 48, nB + 24);
    }
    // compute on current stage: 4 A frags, stream 4 B frags, 16 WMMAs
    v16bf af[4];
    for (int i = 0; i < 4; ++i)
      af[i] = load_frag_at(&As[cur][(wm + i * 16 + l16) * 32], 0, half);
    for (int j = 0; j < 4; ++j) {
      v16bf bfm = load_frag_at(&Bs[cur][(wn + j * 16 + l16) * 32], 0, half);
      for (int i = 0; i < 4; ++i)
        acc[i][j] = wmma_bf16(af[i], bfm, acc[i][j]);
    }
    wait_async0();
    __syncthreads();
  }

  // Epilogue. C layout: lanes 0-15 -> M=r, lanes 16-31 -> M=r+8; N = l16.
  for (int i = 0; i < 4; ++i)
    for (int j = 0; j < 4; ++j)
      for (int r = 0; r < 8; ++r) {
        int m = bm + wm + i * 16 + r + half * 8;
        int n = bn + wn + j * 16 + l16;
        float val = acc[i][j][r] + bias[n];
        if (MODE == 0) {
          // m = b*2048 + l ; n = h*128 + d  -> dst[(b*16+h)*2048*128 + l*128 + d]
          int b = m >> 11, l = m & 2047;
          int h = n >> 7,  d = n & 127;
          out_qkv[(((size_t)b * 16 + h) * 2048 + l) * 128 + d] = (bf16)val;
        } else {
          out_f32[(size_t)m * N + n] = val;
        }
      }
}

// ---------------------------------------------------------------------------
// Stage 2: sliding-window flash attention.
// Grid: (L/128, B*H). Block 256 = 8 waves; wave w owns 16 queries.
// Key chunks of 32 stream through LDS (K async row-major, V transposed).
// ---------------------------------------------------------------------------
__global__ __launch_bounds__(256) void attn_kernel(
    const bf16* __restrict__ Q, const bf16* __restrict__ Kb,
    const bf16* __restrict__ Vb, bf16* __restrict__ AO) {
  __shared__ bf16 Ks[32 * 128];   // [key][d]
  __shared__ bf16 Vt[128 * 32];   // [d][key]  (transposed V chunk)
  __shared__ bf16 Ps[8][16 * 32]; // per-wave P staging (C-layout -> A-layout)

  const int tid  = threadIdx.x;
  const int lane = tid & 31;
  const int wid  = tid >> 5;
  const int half = lane >> 4;
  const int l16  = lane & 15;
  const int qblock = blockIdx.x * 128;
  const int bh = blockIdx.y;            // b*16 + h
  const int b  = bh >> 4, h = bh & 15;
  const size_t head_off = (size_t)bh * 2048 * 128;
  const int qb = qblock + wid * 16;     // this wave's first query row
  const float scale = 0.08838834764831845f; // 1/sqrt(128)

  // Q fragments for this wave's 16 rows (registers; 4 K-slices of D=128)
  v16bf qf[4];
  {
    const bf16* qrow = Q + head_off + (size_t)(qb + l16) * 128;
    for (int c = 0; c < 4; ++c) qf[c] = load_frag_at(qrow, c * 32, half);
  }

  v8f oacc[8];
  for (int t = 0; t < 8; ++t)
    for (int e = 0; e < 8; ++e) oacc[t][e] = 0.0f;
  float mrow[8], lrow[8];
  for (int r = 0; r < 8; ++r) { mrow[r] = -1.0e30f; lrow[r] = 0.0f; }

  int kc0 = qblock - 255;
  if (kc0 < 0) kc0 = 0;
  kc0 &= ~31;
  const int kend = qblock + 127;

  const int lkey = tid >> 3;          // 0..31
  const int dseg = (tid & 7) * 16;    // 0..112
  const unsigned ksOff = lds_off_of(&Ks[lkey * 128 + dseg]);

  for (int kc = kc0; kc <= kend; kc += 32) {
    __syncthreads();
    {   // K chunk via async global->LDS; V chunk transposed via registers
      const bf16* krow = Kb + head_off + (size_t)(kc + lkey) * 128 + dseg;
      async_b128(ksOff,      krow);
      async_b128(ksOff + 16, krow + 8);
      union { u32x4 q[2]; bf16 e[16]; } vv;
      const bf16* vrow = Vb + head_off + (size_t)(kc + lkey) * 128 + dseg;
      vv.q[0] = reinterpret_cast<const u32x4*>(vrow)[0];
      vv.q[1] = reinterpret_cast<const u32x4*>(vrow)[1];
      for (int e = 0; e < 16; ++e) Vt[(dseg + e) * 32 + lkey] = vv.e[e];
    }
    wait_async0();
    __syncthreads();

    // skip chunks entirely outside this wave's window [qb-255, qb+15]
    if (kc + 31 < qb - 255 || kc > qb + 15) continue;

    // S = Q * K^T for 32 keys: two 16x16 tiles, contracted over D in 4 steps
    v8f s0, s1;
    for (int e = 0; e < 8; ++e) { s0[e] = 0.0f; s1[e] = 0.0f; }
    for (int c = 0; c < 4; ++c) {
      v16bf kf0 = load_frag_at(&Ks[(l16) * 128],      c * 32, half);
      v16bf kf1 = load_frag_at(&Ks[(16 + l16) * 128], c * 32, half);
      s0 = wmma_bf16(qf[c], kf0, s0);
      s1 = wmma_bf16(qf[c], kf1, s1);
    }

    // mask + scale + online softmax (row reductions across 16-lane halves)
    float p0[8], p1[8];
    const int j0 = kc + l16, j1 = kc + 16 + l16;
    for (int r = 0; r < 8; ++r) {
      const int i = qb + r + half * 8;
      float a0 = s0[r] * scale;
      float a1 = s1[r] * scale;
      if (j0 > i || j0 < i - 255) a0 = -1.0e30f;
      if (j1 > i || j1 < i - 255) a1 = -1.0e30f;
      float v = fmaxf(a0, a1);
      for (int d = 1; d < 16; d <<= 1) v = fmaxf(v, __shfl_xor(v, d, 32));
      const float mnew = fmaxf(mrow[r], v);
      const float f = __expf(mrow[r] - mnew);
      const float e0 = __expf(a0 - mnew);
      const float e1 = __expf(a1 - mnew);
      float rs = e0 + e1;
      for (int d = 1; d < 16; d <<= 1) rs += __shfl_xor(rs, d, 32);
      lrow[r] = lrow[r] * f + rs;
      mrow[r] = mnew;
      for (int t = 0; t < 8; ++t) oacc[t][r] *= f;
      p0[r] = e0; p1[r] = e1;
    }

    // C-layout P -> A-layout via per-wave LDS staging
    bf16* ps = &Ps[wid][0];
    for (int r = 0; r < 8; ++r) {
      const int row = r + half * 8;
      ps[row * 32 + l16]      = (bf16)p0[r];
      ps[row * 32 + 16 + l16] = (bf16)p1[r];
    }
    v16bf pf = load_frag_at(ps + l16 * 32, 0, half);

    // O += P(16x32) * V(32x128): 8 d-tiles from transposed V
    for (int t = 0; t < 8; ++t) {
      v16bf vf = load_frag_at(&Vt[(t * 16 + l16) * 32], 0, half);
      oacc[t] = wmma_bf16(pf, vf, oacc[t]);
    }
  }

  // normalize and store to (B, L, C) bf16 attention output
  for (int r = 0; r < 8; ++r) {
    const float inv = 1.0f / lrow[r];
    const int l = qb + r + half * 8;
    bf16* dst = AO + ((size_t)b * 2048 + l) * 2048 + h * 128;
    for (int t = 0; t < 8; ++t)
      dst[t * 16 + l16] = (bf16)(oacc[t][r] * inv);
  }
}

// ---------------------------------------------------------------------------
// Host launcher
// ---------------------------------------------------------------------------
extern "C" void kernel_launch(void* const* d_in, const int* in_sizes, int n_in,
                              void* d_out, int out_size, void* d_ws, size_t ws_size,
                              hipStream_t stream) {
  (void)in_sizes; (void)n_in; (void)out_size; (void)ws_size;
  const float* x  = (const float*)d_in[0];
  const float* Wq = (const float*)d_in[1];
  const float* bq = (const float*)d_in[2];
  const float* Wk = (const float*)d_in[3];
  const float* bk = (const float*)d_in[4];
  const float* Wv = (const float*)d_in[5];
  const float* bv = (const float*)d_in[6];
  const float* Wo = (const float*)d_in[7];
  const float* bo = (const float*)d_in[8];
  float* out = (float*)d_out;

  const int Bsz = 2, L = 2048, C = 2048;
  const int M = Bsz * L;            // 4096
  const size_t XC = (size_t)M * C;  // 8,388,608
  const size_t CC = (size_t)C * C;  // 4,194,304

  bf16* xb   = (bf16*)d_ws;
  bf16* wqb  = xb   + XC;
  bf16* wkb  = wqb  + CC;
  bf16* wvb  = wkb  + CC;
  bf16* wob  = wvb  + CC;
  bf16* qbuf = wob  + CC;
  bf16* kbuf = qbuf + XC;
  bf16* vbuf = kbuf + XC;
  bf16* aob  = vbuf + XC;           // total ~112 MB of bf16

  // Stage 0: convert to bf16
  cvt_f32_bf16<<<(int)(XC / 4 / 256), 256, 0, stream>>>(x,  xb,  (int)XC);
  cvt_f32_bf16<<<(int)(CC / 4 / 256), 256, 0, stream>>>(Wq, wqb, (int)CC);
  cvt_f32_bf16<<<(int)(CC / 4 / 256), 256, 0, stream>>>(Wk, wkb, (int)CC);
  cvt_f32_bf16<<<(int)(CC / 4 / 256), 256, 0, stream>>>(Wv, wvb, (int)CC);
  cvt_f32_bf16<<<(int)(CC / 4 / 256), 256, 0, stream>>>(Wo, wob, (int)CC);

  // Stage 1: QKV projections -> (B,H,L,D) bf16
  dim3 ggrid(C / 256, M / 128);     // (8, 32)
  gemm_bf16_kernel<0><<<ggrid, 256, 0, stream>>>(xb, wqb, bq, qbuf, nullptr, M, C, C);
  gemm_bf16_kernel<0><<<ggrid, 256, 0, stream>>>(xb, wkb, bk, kbuf, nullptr, M, C, C);
  gemm_bf16_kernel<0><<<ggrid, 256, 0, stream>>>(xb, wvb, bv, vbuf, nullptr, M, C, C);

  // Stage 2: sliding-window attention -> (B,L,C) bf16
  dim3 agrid(L / 128, Bsz * 16);    // (16, 32)
  attn_kernel<<<agrid, 256, 0, stream>>>(qbuf, kbuf, vbuf, aob);

  // Stage 3: output projection -> f32
  gemm_bf16_kernel<1><<<ggrid, 256, 0, stream>>>(aob, wob, bo, nullptr, out, M, C, C);
}
